// EvoformerBlock_57501022158908
// MI455X (gfx1250) — matrix operand order repaired and checked by reading, not compiled, not run on hardware
//
#include <hip/hip_runtime.h>
#include <math.h>

// ---------------------------------------------------------------------------
// Types for CDNA5 WMMA (wave32): v_wmma_f32_16x16x32_f16
// ---------------------------------------------------------------------------
typedef _Float16 v16h __attribute__((ext_vector_type(16)));
typedef _Float16 h8   __attribute__((ext_vector_type(8)));
typedef _Float16 h4   __attribute__((ext_vector_type(4)));
typedef float    v8f  __attribute__((ext_vector_type(8)));
typedef float    f4   __attribute__((ext_vector_type(4)));

static __device__ inline v8f zero8() {
  v8f z;
#pragma unroll
  for (int i = 0; i < 8; i++) z[i] = 0.0f;
  return z;
}

static __device__ inline v16h pack16(h8 lo, h8 hi) {
  v16h r;
#pragma unroll
  for (int i = 0; i < 8; i++) { r[i] = lo[i]; r[i + 8] = hi[i]; }
  return r;
}

#define WMMA16(a, b, c) \
  __builtin_amdgcn_wmma_f32_16x16x32_f16(false, (a), false, (b), (short)0, (c), false, false)

// Problem constants
#define S_  128
#define R_  256
#define CM_ 256
#define CZ_ 128
#define SR_ (S_ * R_)          // 32768
#define RR_ (R_ * R_)          // 65536

// ---------------------------------------------------------------------------
// LayerNorm over last dim (C = 128 or 256). One block (128 thr) per row.
// ---------------------------------------------------------------------------
__global__ __launch_bounds__(128) void ev_ln(const float* __restrict__ x,
                                             const float* __restrict__ g,
                                             const float* __restrict__ b,
                                             float* __restrict__ y, int C) {
  __shared__ float red[128];
  const size_t row = blockIdx.x;
  const float* xr = x + row * (size_t)C;
  float* yr = y + row * (size_t)C;
  int tid = threadIdx.x;
  float s = 0.f, s2 = 0.f;
  for (int c = tid; c < C; c += 128) { float v = xr[c]; s += v; s2 += v * v; }
  red[tid] = s; __syncthreads();
  for (int o = 64; o > 0; o >>= 1) { if (tid < o) red[tid] += red[tid + o]; __syncthreads(); }
  float mean = red[0] / C;
  __syncthreads();
  red[tid] = s2; __syncthreads();
  for (int o = 64; o > 0; o >>= 1) { if (tid < o) red[tid] += red[tid + o]; __syncthreads(); }
  float var = red[0] / C - mean * mean;
  float inv = rsqrtf(var + 1e-5f);
  for (int c = tid; c < C; c += 128) yr[c] = (xr[c] - mean) * inv * g[c] + b[c];
}

// ---------------------------------------------------------------------------
// Generic WMMA GEMM:  C = resid + act(A@B(^T) + bias) * rowscale
// A: (M,K) row-major f32.  B: (K,N) row-major, or (N,K) if TRB.
// Batched via blockIdx.z with element strides sA,sB,sC.
// Requires M % 64 == 0, K % 32 == 0, N % 4 == 0 (holds for all uses here).
// ACT: 0=none 1=relu 2=sigmoid
// ---------------------------------------------------------------------------
template <int ACT, int TRB>
__global__ __launch_bounds__(128) void ev_gemm(const float* __restrict__ A,
                                               const float* __restrict__ B,
                                               const float* __restrict__ bias,
                                               const float* __restrict__ resid,
                                               const float* __restrict__ rowscale,
                                               float* __restrict__ C,
                                               int M, int N, int K,
                                               long long sA, long long sB, long long sC) {
  __shared__ __align__(16) _Float16 As[64][32];
  __shared__ __align__(16) _Float16 Bs[64][32];  // Bs[n][k]
  const int bz = blockIdx.z;
  const float* Ab = A + (long long)bz * sA;
  const float* Bb = B + (long long)bz * sB;
  float* Cb = C + (long long)bz * sC;
  const float* Rb = resid ? resid + (long long)bz * sC : (const float*)0;
  const int m0 = blockIdx.y * 64, n0 = blockIdx.x * 64;
  const int tid = threadIdx.x;
  const int wv = tid >> 5, lane = tid & 31;
  const int wm = (wv >> 1) * 32, wn = (wv & 1) * 32;
  const int ml = lane & 15, hig = lane >> 4;
  v8f acc[2][2];
  acc[0][0] = zero8(); acc[0][1] = zero8(); acc[1][0] = zero8(); acc[1][1] = zero8();

  for (int k0 = 0; k0 < K; k0 += 32) {
    __syncthreads();
    // A tile: 64x32 f32 -> f16, vectorized 128-bit loads, 64-bit LDS stores
#pragma unroll
    for (int it = 0; it < 4; it++) {
      int idx = tid + it * 128;                 // 512 float4 groups
      int r = idx >> 3, c4 = (idx & 7) * 4;
      f4 v = *(const f4*)(Ab + (size_t)(m0 + r) * K + (k0 + c4));
      h4 hv;
#pragma unroll
      for (int i = 0; i < 4; i++) hv[i] = (_Float16)v[i];
      *(h4*)(&As[r][c4]) = hv;
    }
    // B tile -> Bs[n][k]
    if (TRB) {
#pragma unroll
      for (int it = 0; it < 4; it++) {
        int idx = tid + it * 128;
        int n = idx >> 3, c4 = (idx & 7) * 4;
        f4 v = {0.f, 0.f, 0.f, 0.f};
        if (n0 + n < N) v = *(const f4*)(Bb + (size_t)(n0 + n) * K + (k0 + c4));
        h4 hv;
#pragma unroll
        for (int i = 0; i < 4; i++) hv[i] = (_Float16)v[i];
        *(h4*)(&Bs[n][c4]) = hv;
      }
    } else {
#pragma unroll
      for (int it = 0; it < 4; it++) {
        int idx = tid + it * 128;
        int c = idx >> 4, n4 = (idx & 15) * 4;  // c: k-row, n4: n offset
        f4 v = {0.f, 0.f, 0.f, 0.f};
        if (n0 + n4 < N) v = *(const f4*)(Bb + (size_t)(k0 + c) * N + (n0 + n4));
#pragma unroll
        for (int i = 0; i < 4; i++) Bs[n4 + i][c] = (_Float16)v[i];
      }
    }
    __syncthreads();
#pragma unroll
    for (int mt = 0; mt < 2; mt++) {
      v16h af = pack16(*(const h8*)(&As[wm + mt * 16 + ml][hig * 16]),
                       *(const h8*)(&As[wm + mt * 16 + ml][hig * 16 + 8]));
#pragma unroll
      for (int nt = 0; nt < 2; nt++) {
        v16h bf = pack16(*(const h8*)(&Bs[wn + nt * 16 + ml][hig * 16]),
                         *(const h8*)(&Bs[wn + nt * 16 + ml][hig * 16 + 8]));
        acc[mt][nt] = WMMA16(af, bf, acc[mt][nt]);
      }
    }
  }
#pragma unroll
  for (int mt = 0; mt < 2; mt++)
#pragma unroll
    for (int nt = 0; nt < 2; nt++)
#pragma unroll
      for (int v = 0; v < 8; v++) {
        int m = m0 + wm + mt * 16 + v + 8 * hig;
        int n = n0 + wn + nt * 16 + ml;
        if (n < N) {
          float x = acc[mt][nt][v];
          if (bias) x += bias[n];
          if (ACT == 1) x = fmaxf(x, 0.0f);
          if (ACT == 2) x = 1.0f / (1.0f + __expf(-x));
          if (rowscale) x *= rowscale[m];
          if (Rb) x += Rb[(size_t)m * N + n];
          Cb[(size_t)m * N + n] = x;
        }
      }
}

// ---------------------------------------------------------------------------
// Small-N GEMM (N = 4 or 8): C[m,n] = sum_k A[m,k] B[k,n]
// ---------------------------------------------------------------------------
__global__ void ev_gemm_small(const float* __restrict__ A, const float* __restrict__ B,
                              float* __restrict__ C, int N, int K, long long total) {
  long long idx = (long long)blockIdx.x * blockDim.x + threadIdx.x;
  if (idx >= total) return;
  int n = (int)(idx % N);
  long long m = idx / N;
  float s = 0.f;
  for (int k = 0; k < K; k++) s += A[m * K + k] * B[(long long)k * N + n];
  C[idx] = s;
}

// ---------------------------------------------------------------------------
// Fused gated attention, one wave per (batch, 16-query tile).
// MODE 0: MSA row  (R/16 x H_M x S), keys=R, bias zb[h,i,j], mask msa[s,j]
// MODE 1: MSA col  (S/16 x H_M x R), keys=S, no bias,        mask msa[t,i]
// MODE 2: triangle (R/16 x H_P x R), keys=R, bias tb[h,j,k], mask pm[i,k]
// Q/K/V/G buffers laid out (rows, nheads*32). G pre-sigmoided. Scale folded
// into Q. Output written gated.
// ---------------------------------------------------------------------------
template <int MODE>
__global__ __launch_bounds__(32) void ev_attn(const float* __restrict__ Q,
                                              const float* __restrict__ K,
                                              const float* __restrict__ V,
                                              const float* __restrict__ G,
                                              float* __restrict__ O,
                                              const float* __restrict__ Bias,
                                              const float* __restrict__ Mask) {
  constexpr int CD = (MODE == 2) ? CZ_ : CM_;
  const int lane = threadIdx.x;
  size_t qoff, koff;
  int qrs, krs, Lk, mstep;
  const float* bias_base = (const float*)0;
  const float* mask_base;
  if constexpr (MODE == 0) {
    int it = blockIdx.x, h = blockIdx.y, s = blockIdx.z;
    qoff = ((size_t)s * R_ + it * 16) * CD + h * 32; qrs = CD;
    koff = (size_t)s * R_ * CD + h * 32;             krs = CD;
    Lk = R_;
    bias_base = Bias + (size_t)h * R_ * R_ + (size_t)it * 16 * R_;
    mask_base = Mask + (size_t)s * R_; mstep = 1;
  } else if constexpr (MODE == 1) {
    int st = blockIdx.x, h = blockIdx.y, i = blockIdx.z;
    qoff = ((size_t)st * 16) * R_ * CD + (size_t)i * CD + h * 32; qrs = R_ * CD;
    koff = (size_t)i * CD + h * 32;                               krs = R_ * CD;
    Lk = S_;
    mask_base = Mask + i; mstep = R_;
  } else {
    int jt = blockIdx.x, h = blockIdx.y, i = blockIdx.z;
    qoff = ((size_t)i * R_ + jt * 16) * CD + h * 32; qrs = CD;
    koff = (size_t)i * R_ * CD + h * 32;             krs = CD;
    Lk = R_;
    bias_base = Bias + (size_t)h * R_ * R_ + (size_t)jt * 16 * R_;
    mask_base = Mask + (size_t)i * R_; mstep = 1;
  }
  const float* qb = Q + qoff;
  const float* kb = K + koff;
  const float* vb = V + koff;
  const float* gb = G + qoff;
  float* ob = O + qoff;

  __shared__ __align__(16) _Float16 Qs[16][32];
  __shared__ __align__(16) _Float16 Ps[16][256];
  __shared__ __align__(16) _Float16 Vst[32][32];  // transposed V tile [c][k]
  __shared__ float Ls[16][256];

  // Q tile (scale folded), vectorized
#pragma unroll
  for (int it = 0; it < 4; it++) {
    int idx = lane + it * 32;              // 128 float4 groups over 16x32
    int m = idx >> 3, c4 = (idx & 7) * 4;
    f4 v = *(const f4*)(qb + (size_t)m * qrs + c4);
    h4 hv;
#pragma unroll
    for (int i = 0; i < 4; i++) hv[i] = (_Float16)(v[i] * 0.17677669529663687f);
    *(h4*)(&Qs[m][c4]) = hv;
  }
  __syncthreads();

  const int ml = lane & 15, hig = lane >> 4;
  v16h aq = pack16(*(const h8*)(&Qs[ml][hig * 16]),
                   *(const h8*)(&Qs[ml][hig * 16 + 8]));

  for (int jt = 0; jt < Lk / 16; jt++) {
    int kj = jt * 16 + ml;
    const float* kp = kb + (size_t)kj * krs + hig * 16;
    v16h bf;
#pragma unroll
    for (int q4 = 0; q4 < 4; q4++) {
      f4 v = *(const f4*)(kp + q4 * 4);
#pragma unroll
      for (int i = 0; i < 4; i++) bf[q4 * 4 + i] = (_Float16)v[i];
    }
    v8f lz = zero8();
    lz = WMMA16(aq, bf, lz);
    float mk = (mask_base[(size_t)kj * mstep] - 1.0f) * 1e9f;
#pragma unroll
    for (int v = 0; v < 8; v++) {
      int m = v + 8 * hig;
      float lg = lz[v] + mk;
      if constexpr (MODE != 1) lg += bias_base[(size_t)m * R_ + kj];
      Ls[m][kj] = lg;
    }
  }
  __syncthreads();

  if (lane < 16) {
    float mx = -1e30f;
    for (int j = 0; j < Lk; j++) mx = fmaxf(mx, Ls[lane][j]);
    float sm = 0.f;
    for (int j = 0; j < Lk; j++) { float e = __expf(Ls[lane][j] - mx); sm += e; Ls[lane][j] = e; }
    float inv = 1.0f / sm;
    for (int j = 0; j < Lk; j++) Ps[lane][j] = (_Float16)(Ls[lane][j] * inv);
  }
  __syncthreads();

  v8f acc0 = zero8(), acc1 = zero8();
  for (int kt = 0; kt < Lk / 32; kt++) {
    // stage V tile (32 keys x 32 ch) transposed into LDS, vectorized loads
    __syncthreads();
#pragma unroll
    for (int it = 0; it < 8; it++) {
      int idx = lane + it * 32;            // 256 float4 groups over 32x32
      int r = idx >> 3, c4 = (idx & 7) * 4;
      f4 v = *(const f4*)(vb + (size_t)(kt * 32 + r) * krs + c4);
#pragma unroll
      for (int i = 0; i < 4; i++) Vst[c4 + i][r] = (_Float16)v[i];
    }
    __syncthreads();

    v16h ap = pack16(*(const h8*)(&Ps[ml][kt * 32 + hig * 16]),
                     *(const h8*)(&Ps[ml][kt * 32 + hig * 16 + 8]));
    v16h bv0 = pack16(*(const h8*)(&Vst[ml][hig * 16]),
                      *(const h8*)(&Vst[ml][hig * 16 + 8]));
    v16h bv1 = pack16(*(const h8*)(&Vst[16 + ml][hig * 16]),
                      *(const h8*)(&Vst[16 + ml][hig * 16 + 8]));
    acc0 = WMMA16(ap, bv0, acc0);
    acc1 = WMMA16(ap, bv1, acc1);
  }
#pragma unroll
  for (int v = 0; v < 8; v++) {
    int m = v + 8 * hig;
    size_t o0 = (size_t)m * qrs + ml;
    size_t o1 = (size_t)m * qrs + 16 + ml;
    ob[o0] = acc0[v] * gb[o0];
    ob[o1] = acc1[v] * gb[o1];
  }
}

// ---------------------------------------------------------------------------
// Generic permutes / elementwise helpers
// ---------------------------------------------------------------------------
__global__ void ev_permute3(const float* __restrict__ in, float* __restrict__ out,
                            int D1, int D2, long long s0, long long s1, long long s2,
                            long long n, int acc) {
  long long idx = (long long)blockIdx.x * blockDim.x + threadIdx.x;
  if (idx >= n) return;
  int i2 = (int)(idx % D2);
  long long t = idx / D2;
  int i1 = (int)(t % D1);
  long long i0 = t / D1;
  long long o = i0 * s0 + (long long)i1 * s1 + (long long)i2 * s2;
  if (acc) out[o] += in[idx]; else out[o] = in[idx];
}

__global__ void ev_permute4(const float* __restrict__ in, float* __restrict__ out,
                            int D1, int D2, int D3,
                            long long s0, long long s1, long long s2, long long s3,
                            long long n) {
  long long idx = (long long)blockIdx.x * blockDim.x + threadIdx.x;
  if (idx >= n) return;
  int i3 = (int)(idx % D3);
  long long t = idx / D3;
  int i2 = (int)(t % D2);
  t /= D2;
  int i1 = (int)(t % D1);
  long long i0 = t / D1;
  out[i0 * s0 + (long long)i1 * s1 + (long long)i2 * s2 + (long long)i3 * s3] = in[idx];
}

__global__ void ev_mask_mul(float* __restrict__ a, const float* __restrict__ mask,
                            int C, long long n) {
  long long i = (long long)blockIdx.x * blockDim.x + threadIdx.x;
  if (i < n) a[i] *= mask[i / C];
}

// dst = mask * sigmoid(pre) * dst
__global__ void ev_gate2(float* __restrict__ dst, const float* __restrict__ pre,
                         const float* __restrict__ mask, int C, long long n) {
  long long i = (long long)blockIdx.x * blockDim.x + threadIdx.x;
  if (i < n) dst[i] = mask[i / C] * (1.0f / (1.0f + __expf(-pre[i]))) * dst[i];
}

__global__ void ev_fma_add(float* __restrict__ z, const float* __restrict__ g,
                           const float* __restrict__ y, long long n) {
  long long i = (long long)blockIdx.x * blockDim.x + threadIdx.x;
  if (i < n) z[i] += g[i] * y[i];
}

__global__ void ev_norm_inv(const float* __restrict__ msk, float* __restrict__ out) {
  int idx = blockIdx.x * blockDim.x + threadIdx.x;
  if (idx >= RR_) return;
  int i = idx >> 8, j = idx & 255;
  float s = 0.f;
  for (int t = 0; t < S_; t++) s += msk[t * R_ + i] * msk[t * R_ + j];
  out[idx] = 1.0f / (s + 1e-3f);
}

// ---------------------------------------------------------------------------
// Host-side dispatch helpers
// ---------------------------------------------------------------------------
static void launch_gemm(hipStream_t st, int act, int trb,
                        const float* A, const float* B, const float* bias,
                        const float* resid, const float* rowscale, float* C,
                        int M, int N, int K,
                        int batch = 1, long long sA = 0, long long sB = 0, long long sC = 0) {
  dim3 g((N + 63) / 64, (M + 63) / 64, batch), b(128);
  if (trb == 0) {
    if (act == 0)      ev_gemm<0, 0><<<g, b, 0, st>>>(A, B, bias, resid, rowscale, C, M, N, K, sA, sB, sC);
    else if (act == 1) ev_gemm<1, 0><<<g, b, 0, st>>>(A, B, bias, resid, rowscale, C, M, N, K, sA, sB, sC);
    else               ev_gemm<2, 0><<<g, b, 0, st>>>(A, B, bias, resid, rowscale, C, M, N, K, sA, sB, sC);
  } else {
    if (act == 0)      ev_gemm<0, 1><<<g, b, 0, st>>>(A, B, bias, resid, rowscale, C, M, N, K, sA, sB, sC);
    else if (act == 1) ev_gemm<1, 1><<<g, b, 0, st>>>(A, B, bias, resid, rowscale, C, M, N, K, sA, sB, sC);
    else               ev_gemm<2, 1><<<g, b, 0, st>>>(A, B, bias, resid, rowscale, C, M, N, K, sA, sB, sC);
  }
}

static void launch_p3(hipStream_t st, const float* in, float* out, int D1, int D2,
                      long long s0, long long s1, long long s2, long long n, int acc = 0) {
  ev_permute3<<<(unsigned)((n + 255) / 256), 256, 0, st>>>(in, out, D1, D2, s0, s1, s2, n, acc);
}

// Parameter indices: inputs m,z,msa_mask,pair_mask then params flattened in
// jax pytree order (nested dicts, keys sorted alphabetically).
enum {
  IN_M = 0, IN_Z, IN_MSAMASK, IN_PAIRMASK,
  COL_BG, COL_BO, COL_LNB, COL_LNG, COL_WG, COL_WK, COL_WO, COL_WQ, COL_WV,          // col
  MT_B1, MT_B2, MT_LNB, MT_LNG, MT_W1, MT_W2,                                        // msa_trans
  OPM_BA, OPM_BB, OPM_BO, OPM_LNB, OPM_LNG, OPM_WA, OPM_WB, OPM_WO,                  // opm
  PT_B1, PT_B2, PT_LNB, PT_LNG, PT_W1, PT_W2,                                        // pair_trans
  ROW_BG, ROW_BO, ROW_LNB, ROW_LNG, ROW_LNZB, ROW_LNZG, ROW_WG, ROW_WK, ROW_WO, ROW_WQ, ROW_WV, ROW_WZB, // row
  TE_BG, TE_BO, TE_LNB, TE_LNG, TE_WB2, TE_WG, TE_WK, TE_WO, TE_WQ, TE_WV,           // tri_end
  TI_BAG, TI_BBG, TI_BG, TI_BO, TI_LN2B, TI_LN2G, TI_LNB, TI_LNG, TI_WA, TI_WAG, TI_WB, TI_WBG, TI_WG, TI_WO, // tri_in
  TO_BAG, TO_BBG, TO_BG, TO_BO, TO_LN2B, TO_LN2G, TO_LNB, TO_LNG, TO_WA, TO_WAG, TO_WB, TO_WBG, TO_WG, TO_WO, // tri_out
  TS_BG, TS_BO, TS_LNB, TS_LNG, TS_WB2, TS_WG, TS_WK, TS_WO, TS_WQ, TS_WV            // tri_start
};

extern "C" void kernel_launch(void* const* d_in, const int* in_sizes, int n_in,
                              void* d_out, int out_size, void* d_ws, size_t ws_size,
                              hipStream_t stream) {
  (void)in_sizes; (void)n_in; (void)out_size; (void)ws_size;
  auto F = [&](int i) { return (const float*)d_in[i]; };
  const float* msa_mask = F(IN_MSAMASK);
  const float* pair_mask = F(IN_PAIRMASK);

  const size_t NM = (size_t)SR_ * CM_;   // 8,388,608
  const size_t NZ = (size_t)RR_ * CZ_;   // 8,388,608
  float* wsf  = (float*)d_ws;
  float* Mcur = wsf;
  float* Zcur = Mcur + NM;
  float* X1   = Zcur + NZ;
  float* X2   = X1 + NM;
  float* Qb   = X2 + NM;
  float* Kb   = Qb + NM;
  float* Vb   = Kb + NM;
  float* Gb   = Vb + NM;
  float* Ob   = Gb + NM;
  float* H1   = Ob + NM;                   // 4*NM floats (transition hidden etc.)
  float* BiasB = H1 + 4 * NM;              // 524288
  float* TmpS  = BiasB + 524288;           // 524288
  float* NormI = TmpS + 524288;            // 65536
  float* MaskT = NormI + 65536;            // 65536

  hipMemcpyAsync(Mcur, d_in[IN_M], NM * sizeof(float), hipMemcpyDeviceToDevice, stream);
  hipMemcpyAsync(Zcur, d_in[IN_Z], NZ * sizeof(float), hipMemcpyDeviceToDevice, stream);

  auto LN = [&](const float* x, const float* g, const float* b, float* y, int rows, int C) {
    ev_ln<<<rows, 128, 0, stream>>>(x, g, b, y, C);
  };

  // ================= MSA row attention =================
  LN(Mcur, F(ROW_LNG), F(ROW_LNB), X1, SR_, CM_);
  launch_gemm(stream, 0, 0, X1, F(ROW_WQ), 0, 0, 0, Qb, SR_, 256, 256);
  launch_gemm(stream, 0, 0, X1, F(ROW_WK), 0, 0, 0, Kb, SR_, 256, 256);
  launch_gemm(stream, 0, 0, X1, F(ROW_WV), 0, 0, 0, Vb, SR_, 256, 256);
  launch_gemm(stream, 2, 0, X1, F(ROW_WG), F(ROW_BG), 0, 0, Gb, SR_, 256, 256);
  LN(Zcur, F(ROW_LNZG), F(ROW_LNZB), X2, RR_, CZ_);
  ev_gemm_small<<<(RR_ * 8 + 255) / 256, 256, 0, stream>>>(X2, F(ROW_WZB), TmpS, 8, 128, (long long)RR_ * 8);
  launch_p3(stream, TmpS, BiasB, R_, 8, R_, 1, RR_, (long long)RR_ * 8);  // (i,j,h)->(h,i,j)
  ev_attn<0><<<dim3(16, 8, 128), 32, 0, stream>>>(Qb, Kb, Vb, Gb, Ob, BiasB, msa_mask);
  launch_gemm(stream, 0, 0, Ob, F(ROW_WO), F(ROW_BO), Mcur, 0, Mcur, SR_, 256, 256);

  // ================= MSA column attention =================
  LN(Mcur, F(COL_LNG), F(COL_LNB), X1, SR_, CM_);
  launch_gemm(stream, 0, 0, X1, F(COL_WQ), 0, 0, 0, Qb, SR_, 256, 256);
  launch_gemm(stream, 0, 0, X1, F(COL_WK), 0, 0, 0, Kb, SR_, 256, 256);
  launch_gemm(stream, 0, 0, X1, F(COL_WV), 0, 0, 0, Vb, SR_, 256, 256);
  launch_gemm(stream, 2, 0, X1, F(COL_WG), F(COL_BG), 0, 0, Gb, SR_, 256, 256);
  ev_attn<1><<<dim3(8, 8, 256), 32, 0, stream>>>(Qb, Kb, Vb, Gb, Ob, (const float*)0, msa_mask);
  launch_gemm(stream, 0, 0, Ob, F(COL_WO), F(COL_BO), Mcur, 0, Mcur, SR_, 256, 256);

  // ================= MSA transition =================
  LN(Mcur, F(MT_LNG), F(MT_LNB), X1, SR_, CM_);
  launch_gemm(stream, 1, 0, X1, F(MT_W1), F(MT_B1), 0, 0, H1, SR_, 1024, 256);
  launch_gemm(stream, 0, 0, H1, F(MT_W2), F(MT_B2), Mcur, 0, Mcur, SR_, 256, 1024);

  // ================= Outer product mean -> z =================
  LN(Mcur, F(OPM_LNG), F(OPM_LNB), X1, SR_, CM_);
  launch_gemm(stream, 0, 0, X1, F(OPM_WA), F(OPM_BA), 0, 0, Qb, SR_, 32, 256);
  ev_mask_mul<<<(SR_ * 32 + 255) / 256, 256, 0, stream>>>(Qb, msa_mask, 32, (long long)SR_ * 32);
  launch_gemm(stream, 0, 0, X1, F(OPM_WB), F(OPM_BB), 0, 0, Kb, SR_, 32, 256);
  ev_mask_mul<<<(SR_ * 32 + 255) / 256, 256, 0, stream>>>(Kb, msa_mask, 32, (long long)SR_ * 32);
  // aT[(i,c), s] = a[s,i,c]
  launch_p3(stream, Qb, Vb, R_, 32, 1, 32LL * S_, S_, (long long)SR_ * 32);
  ev_norm_inv<<<RR_ / 256, 256, 0, stream>>>(msa_mask, NormI);
  for (int ic = 0; ic < 16; ic++) {
    // G[(i',c),(j,d)] = aT_chunk(512x128) @ b(128x8192)
    launch_gemm(stream, 0, 0, Vb + (size_t)ic * 65536, Kb, 0, 0, 0, H1, 512, 8192, 128);
    // (i',c,j,d) -> (i',j,c,d)
    ev_permute4<<<(4194304 + 255) / 256, 256, 0, stream>>>(H1, H1 + 4194304, 32, 256, 32,
                                                           262144LL, 32LL, 1024LL, 1LL, 4194304LL);
    // z += (T @ wo + bo) / norm
    launch_gemm(stream, 0, 0, H1 + 4194304, F(OPM_WO), F(OPM_BO),
                Zcur + (size_t)ic * 524288, NormI + ic * 4096,
                Zcur + (size_t)ic * 524288, 4096, 128, 1024);
  }

  // ================= Triangle multiplication (outgoing, incoming) ==========
  for (int dir = 0; dir < 2; dir++) {
    int WAG = dir == 0 ? TO_WAG : TI_WAG, BAG = dir == 0 ? TO_BAG : TI_BAG;
    int WA = dir == 0 ? TO_WA : TI_WA,   WBG = dir == 0 ? TO_WBG : TI_WBG;
    int BBG = dir == 0 ? TO_BBG : TI_BBG, WB = dir == 0 ? TO_WB : TI_WB;
    int WG = dir == 0 ? TO_WG : TI_WG,   BGI = dir == 0 ? TO_BG : TI_BG;
    int L2G = dir == 0 ? TO_LN2G : TI_LN2G, L2B = dir == 0 ? TO_LN2B : TI_LN2B;
    int WO = dir == 0 ? TO_WO : TI_WO,   BO = dir == 0 ? TO_BO : TI_BO;
    int LG = dir == 0 ? TO_LNG : TI_LNG, LB = dir == 0 ? TO_LNB : TI_LNB;

    LN(Zcur, F(LG), F(LB), X1, RR_, CZ_);
    launch_gemm(stream, 0, 0, X1, F(WAG), F(BAG), 0, 0, X2, RR_, 128, 128);
    launch_gemm(stream, 0, 0, X1, F(WA), 0, 0, 0, Qb, RR_, 128, 128);
    ev_gate2<<<(unsigned)((NZ + 255) / 256), 256, 0, stream>>>(Qb, X2, pair_mask, 128, (long long)NZ);
    launch_gemm(stream, 0, 0, X1, F(WBG), F(BBG), 0, 0, X2, RR_, 128, 128);
    launch_gemm(stream, 0, 0, X1, F(WB), 0, 0, 0, Kb, RR_, 128, 128);
    ev_gate2<<<(unsigned)((NZ + 255) / 256), 256, 0, stream>>>(Kb, X2, pair_mask, 128, (long long)NZ);
    // channel-major: outgoing at[c][i][k]=a[i,k,c]; incoming at[c][i][k]=a[k,i,c]
    if (dir == 0) {
      launch_p3(stream, Qb, H1, R_, 128, R_, 1, RR_, (long long)NZ);
      launch_p3(stream, Kb, H1 + NZ, R_, 128, R_, 1, RR_, (long long)NZ);
    } else {
      launch_p3(stream, Qb, H1, R_, 128, 1, R_, RR_, (long long)NZ);
      launch_p3(stream, Kb, H1 + NZ, R_, 128, 1, R_, RR_, (long long)NZ);
    }
    // batched over c: O_c = A_c @ B_c^T
    launch_gemm(stream, 0, 1, H1, H1 + NZ, 0, 0, 0, X2, 256, 256, 256, 128, 65536, 65536, 65536);
    // (c,i,j) -> (i,j,c)
    launch_p3(stream, X2, Ob, R_, R_, 1, (long long)R_ * 128, 128, (long long)NZ);
    LN(Ob, F(L2G), F(L2B), X2, RR_, CZ_);
    launch_gemm(stream, 0, 0, X2, F(WO), F(BO), 0, 0, Vb, RR_, 128, 128);
    launch_gemm(stream, 2, 0, X1, F(WG), F(BGI), 0, 0, Kb, RR_, 128, 128);
    ev_fma_add<<<(unsigned)((NZ + 255) / 256), 256, 0, stream>>>(Zcur, Kb, Vb, (long long)NZ);
  }

  // ================= Triangle attention (starting node) =================
  LN(Zcur, F(TS_LNG), F(TS_LNB), X1, RR_, CZ_);
  launch_gemm(stream, 0, 0, X1, F(TS_WQ), 0, 0, 0, Qb, RR_, 128, 128);
  launch_gemm(stream, 0, 0, X1, F(TS_WK), 0, 0, 0, Kb, RR_, 128, 128);
  launch_gemm(stream, 0, 0, X1, F(TS_WV), 0, 0, 0, Vb, RR_, 128, 128);
  launch_gemm(stream, 2, 0, X1, F(TS_WG), F(TS_BG), 0, 0, Gb, RR_, 128, 128);
  ev_gemm_small<<<(RR_ * 4 + 255) / 256, 256, 0, stream>>>(X1, F(TS_WB2), TmpS, 4, 128, (long long)RR_ * 4);
  launch_p3(stream, TmpS, BiasB, R_, 4, R_, 1, RR_, (long long)RR_ * 4);  // (j,k,h)->(h,j,k)
  ev_attn<2><<<dim3(16, 4, 256), 32, 0, stream>>>(Qb, Kb, Vb, Gb, Ob, BiasB, pair_mask);
  launch_gemm(stream, 0, 0, Ob, F(TS_WO), F(TS_BO), Zcur, 0, Zcur, RR_, 128, 128);

  // ================= Triangle attention (ending node) =================
  launch_p3(stream, Zcur, X2, R_, 128, 128, (long long)R_ * 128, 1, (long long)NZ);  // z^T
  launch_p3(stream, pair_mask, MaskT, R_, 1, 1, R_, 1, (long long)RR_);              // mask^T
  LN(X2, F(TE_LNG), F(TE_LNB), X1, RR_, CZ_);
  launch_gemm(stream, 0, 0, X1, F(TE_WQ), 0, 0, 0, Qb, RR_, 128, 128);
  launch_gemm(stream, 0, 0, X1, F(TE_WK), 0, 0, 0, Kb, RR_, 128, 128);
  launch_gemm(stream, 0, 0, X1, F(TE_WV), 0, 0, 0, Vb, RR_, 128, 128);
  launch_gemm(stream, 2, 0, X1, F(TE_WG), F(TE_BG), 0, 0, Gb, RR_, 128, 128);
  ev_gemm_small<<<(RR_ * 4 + 255) / 256, 256, 0, stream>>>(X1, F(TE_WB2), TmpS, 4, 128, (long long)RR_ * 4);
  launch_p3(stream, TmpS, BiasB, R_, 4, R_, 1, RR_, (long long)RR_ * 4);
  ev_attn<2><<<dim3(16, 4, 256), 32, 0, stream>>>(Qb, Kb, Vb, Gb, Ob, BiasB, MaskT);
  launch_gemm(stream, 0, 0, Ob, F(TE_WO), F(TE_BO), 0, 0, Kb, RR_, 128, 128);
  // z += transpose01(result)
  launch_p3(stream, Kb, Zcur, R_, 128, 128, (long long)R_ * 128, 1, (long long)NZ, 1);

  // ================= Pair transition =================
  LN(Zcur, F(PT_LNG), F(PT_LNB), X1, RR_, CZ_);
  launch_gemm(stream, 1, 0, X1, F(PT_W1), F(PT_B1), 0, 0, H1, RR_, 512, 128);
  launch_gemm(stream, 0, 0, H1, F(PT_W2), F(PT_B2), Zcur, 0, Zcur, RR_, 128, 512);

  // ================= Outputs =================
  hipMemcpyAsync(d_out, Mcur, NM * sizeof(float), hipMemcpyDeviceToDevice, stream);
  hipMemcpyAsync((float*)d_out + NM, Zcur, NZ * sizeof(float), hipMemcpyDeviceToDevice, stream);
}
